// CausalSelfAttention_2224793059960
// MI455X (gfx1250) — compile-verified, matrix operand order checked by reading, not run on hardware
//
#include <hip/hip_runtime.h>

// ---------------------------------------------------------------------------
// Causal self-attention (GQA) for MI455X / gfx1250, wave32 + WMMA f16.
//   qkv = x @ Wqkv           (WMMA GEMM, TDM-staged tiles, f16 in / f32 acc)
//   rope(q,k); flash attention with online softmax (WMMA for S=QK^T and P*V)
//   out = y @ Wproj          (WMMA GEMM)
// ---------------------------------------------------------------------------

typedef __attribute__((ext_vector_type(4)))  _Float16 v4h;
typedef __attribute__((ext_vector_type(8)))  _Float16 v8h;
typedef __attribute__((ext_vector_type(16))) _Float16 v16h;
typedef __attribute__((ext_vector_type(8)))  float    v8f;

static constexpr int Bq  = 4;
static constexpr int Tq  = 1024;
static constexpr int Cq  = 4096;
static constexpr int Hq  = 32;
static constexpr int KVq = 8;
static constexpr int HDq = 128;
static constexpr int REPq = Hq / KVq;               // 4
static constexpr int NQKV = (Hq + 2 * KVq) * HDq;   // 6144

#define WMMA_F16(a, b, c) \
  __builtin_amdgcn_wmma_f32_16x16x32_f16(false, (a), false, (b), (short)0, (c), false, false)

__device__ __forceinline__ v16h cat8(v8h a, v8h b) {
  return __builtin_shufflevector(a, b, 0, 1, 2, 3, 4, 5, 6, 7, 8, 9, 10, 11, 12, 13, 14, 15);
}

// Wave-local LDS ordering: same-wave DS ops execute in order at the LDS; we
// only need to stop compiler reordering (zero hardware cost).
__device__ __forceinline__ void wave_lds_fence() {
  asm volatile("" ::: "memory");
  __builtin_amdgcn_wave_barrier();
  asm volatile("" ::: "memory");
}

// ---------------------------------------------------------------------------
// Tensor Data Mover (TDM) path — this toolchain exposes the 6-arg builtin
// (u32x4 g0, i32x8 g1, i32x4 g2, i32x4 g3, i32x8 g4, i32 cpol).
// ---------------------------------------------------------------------------
#if defined(__AMDGCN__) && __has_builtin(__builtin_amdgcn_tensor_load_to_lds) && \
    __has_builtin(__builtin_amdgcn_s_wait_tensorcnt)
#define USE_TDM 1
#else
#define USE_TDM 0
#endif

#if USE_TDM
typedef __attribute__((ext_vector_type(4))) unsigned int u32x4;
typedef __attribute__((ext_vector_type(8))) int          i32x8;
typedef __attribute__((ext_vector_type(4))) int          i32x4;

// 2-D tile load: tile_k x tile_rows f16 elements from a row-major tensor with
// row stride `stride_elems`, into LDS at `lds_addr`, inserting 16B of padding
// after every 128B row (=> 72-half LDS row stride to spread banks).
// D# layout per CDNA5 ISA sec. 8.3/8.4.
__device__ __forceinline__ void tdm_load_tile_f16(
    unsigned lds_addr, const _Float16* gptr, unsigned tile_k, unsigned tile_rows,
    unsigned tensor_k, unsigned tensor_rows, unsigned stride_elems) {
  const unsigned long long ga = (unsigned long long)gptr;
  u32x4 g0;
  g0[0] = 1u;                                   // count=1 (valid), user mode
  g0[1] = lds_addr;                             // lds_addr[31:0]
  g0[2] = (unsigned)(ga & 0xffffffffu);         // global_addr[31:0]
  g0[3] = (unsigned)((ga >> 32) & 0x01ffffffu)  // global_addr[56:32]
        | (2u << 30);                           // type = 2 ("image")
  i32x8 g1;
  g1[0] = (1 << 16)                             // data_size = 1 -> 2 bytes
        | (1 << 20)                             // pad_enable
        | (4 << 22)                             // pad_interval: 32 dwords (128B)
        | (3 << 25);                            // pad_amount: 4 dwords (16B)
  g1[1] = (int)((tensor_k & 0xffffu) << 16);                              // tensor_dim0[15:0]
  g1[2] = (int)(((tensor_k >> 16) & 0xffffu) | ((tensor_rows & 0xffffu) << 16));
  g1[3] = (int)(((tensor_rows >> 16) & 0xffffu) | ((tile_k & 0xffffu) << 16)); // tile_dim0
  g1[4] = (int)(tile_rows & 0xffffu);           // tile_dim1 (tile_dim2 = 0 -> 2D)
  g1[5] = (int)stride_elems;                    // tensor_dim0_stride[31:0]
  g1[6] = 0;
  g1[7] = 0;
  const i32x4 gz4 = {0, 0, 0, 0};               // groups 2/3 unused for 2-D
  const i32x8 gz8 = {0, 0, 0, 0, 0, 0, 0, 0};
  __builtin_amdgcn_tensor_load_to_lds(g0, g1, gz4, gz4, gz8, 0);
}

__device__ __forceinline__ unsigned lds_offset_of(const void* p) {
  // Generic LDS addresses carry the LDS byte offset in addr[31:0] (ISA 10.2).
  return (unsigned)(unsigned long long)p;
}
#endif

// ---------------------------------------------------------------------------
// fp32 -> f16 bulk convert (4 elements/thread, b128 reads)
// ---------------------------------------------------------------------------
__global__ __launch_bounds__(256) void f32_to_f16_kernel(
    const float* __restrict__ in, _Float16* __restrict__ out, long n) {
  long i = ((long)blockIdx.x * 256 + threadIdx.x) * 4;
  if (i + 3 < n) {
    float4 v = *(const float4*)(in + i);
    v4h o = {(_Float16)v.x, (_Float16)v.y, (_Float16)v.z, (_Float16)v.w};
    *(v4h*)(out + i) = o;
  }
}

// ---------------------------------------------------------------------------
// fp32 [R, Ncl] -> f16 [Ncl, R]  (LDS-tiled transpose, both sides coalesced)
// ---------------------------------------------------------------------------
__global__ __launch_bounds__(256) void transpose_f32_f16_kernel(
    const float* __restrict__ in, _Float16* __restrict__ out, int R, int Ncl) {
  __shared__ _Float16 tile[32][33];
  const int c0 = blockIdx.x * 32;
  const int r0 = blockIdx.y * 32;
  const int tx = threadIdx.x;   // 0..31
  const int ty = threadIdx.y;   // 0..7
#pragma unroll
  for (int i = 0; i < 32; i += 8)
    tile[ty + i][tx] = (_Float16)in[(long)(r0 + ty + i) * Ncl + c0 + tx];
  __syncthreads();
#pragma unroll
  for (int i = 0; i < 32; i += 8)
    out[(long)(c0 + ty + i) * R + r0 + tx] = tile[tx][ty + i];
}

// ---------------------------------------------------------------------------
// GEMM: C[M,N] (f32) = A[M,K] (f16, row-major) * Bt[N,K] (f16, N-major).
// Workgroup: 256 threads = 8 waves -> 128x128 tile; wave = 64x32 (8 WMMA acc).
// KTILE=64. TDM path: double-buffered tensor_load_to_lds issued by wave 0.
// Fallback: batched global->LDS staging.
// ---------------------------------------------------------------------------
__global__ __launch_bounds__(256) void gemm_f16_wmma(
    const _Float16* __restrict__ A, const _Float16* __restrict__ Bt,
    float* __restrict__ C, int M, int N, int K) {
  constexpr int KT = 64;
  constexpr int LS = 72;  // 64 + 8 halves pad (16B aligned rows, spreads banks)

  const int tid  = threadIdx.x;
  const int wave = tid >> 5;
  const int lane = tid & 31;
  const int lLo  = lane & 15;
  const int lHi  = lane >> 4;
  const int wm   = wave >> 2;  // 0..1 (M)
  const int wn   = wave & 3;   // 0..3 (N)
  const long bN  = (long)blockIdx.x * 128;
  const long bM  = (long)blockIdx.y * 128;

  const v8f zero = {0.f, 0.f, 0.f, 0.f, 0.f, 0.f, 0.f, 0.f};
  v8f acc[4][2];
#pragma unroll
  for (int i = 0; i < 4; ++i)
#pragma unroll
    for (int j = 0; j < 2; ++j) acc[i][j] = zero;

#if USE_TDM
  __shared__ __align__(16) _Float16 sA[2][128 * LS];
  __shared__ __align__(16) _Float16 sB[2][128 * LS];

  if (wave == 0) {  // TDM ignores EXEC; one wave issues the whole-tile DMAs
    tdm_load_tile_f16(lds_offset_of(&sA[0][0]), A + bM * K, KT, 128,
                      (unsigned)K, 128u, (unsigned)K);
    tdm_load_tile_f16(lds_offset_of(&sB[0][0]), Bt + bN * K, KT, 128,
                      (unsigned)K, 128u, (unsigned)K);
  }
  int buf = 0;
  for (int k0 = 0; k0 < K; k0 += KT) {
    if (wave == 0) __builtin_amdgcn_s_wait_tensorcnt((short)0);
    __syncthreads();  // tile `buf` ready; prior reads of `buf^1` all retired
    if (wave == 0 && (k0 + KT) < K) {
      tdm_load_tile_f16(lds_offset_of(&sA[buf ^ 1][0]), A + bM * K + k0 + KT,
                        KT, 128, (unsigned)K, 128u, (unsigned)K);
      tdm_load_tile_f16(lds_offset_of(&sB[buf ^ 1][0]), Bt + bN * K + k0 + KT,
                        KT, 128, (unsigned)K, 128u, (unsigned)K);
    }
    const _Float16* cA = &sA[buf][0];
    const _Float16* cB = &sB[buf][0];
#else
  __shared__ __align__(16) _Float16 sA[128 * LS];
  __shared__ __align__(16) _Float16 sB[128 * LS];

  for (int k0 = 0; k0 < K; k0 += KT) {
    // Batched staging: issue all 8 b128 loads, then all 8 ds stores.
    v8h ra[4], rb[4];
#pragma unroll
    for (int p = 0; p < 4; ++p) {
      const int lin = p * 256 + tid;
      const int row = lin >> 3;
      const int ch  = lin & 7;
      ra[p] = *(const v8h*)(A  + (bM + row) * K + k0 + ch * 8);
      rb[p] = *(const v8h*)(Bt + (bN + row) * K + k0 + ch * 8);
    }
#pragma unroll
    for (int p = 0; p < 4; ++p) {
      const int lin = p * 256 + tid;
      const int row = lin >> 3;
      const int ch  = lin & 7;
      *(v8h*)&sA[row * LS + ch * 8] = ra[p];
      *(v8h*)&sB[row * LS + ch * 8] = rb[p];
    }
    if (k0 + KT < K) {  // pull next K-tile toward L2 (global_prefetch_b8)
      __builtin_prefetch(A  + (bM + (tid >> 1)) * K + k0 + KT + (tid & 1) * 32, 0, 1);
      __builtin_prefetch(Bt + (bN + (tid >> 1)) * K + k0 + KT + (tid & 1) * 32, 0, 1);
    }
    __syncthreads();
    const _Float16* cA = &sA[0];
    const _Float16* cB = &sB[0];
#endif

#pragma unroll
    for (int kk = 0; kk < KT; kk += 32) {
      v16h af[4];
#pragma unroll
      for (int mt = 0; mt < 4; ++mt) {
        const int r = wm * 64 + mt * 16 + lLo;
        af[mt] = cat8(*(const v8h*)&cA[r * LS + kk + lHi * 8],
                      *(const v8h*)&cA[r * LS + kk + 16 + lHi * 8]);
      }
      v16h bf[2];
#pragma unroll
      for (int nt = 0; nt < 2; ++nt) {
        const int r = wn * 32 + nt * 16 + lLo;
        bf[nt] = cat8(*(const v8h*)&cB[r * LS + kk + lHi * 8],
                      *(const v8h*)&cB[r * LS + kk + 16 + lHi * 8]);
      }
#pragma unroll
      for (int mt = 0; mt < 4; ++mt)
#pragma unroll
        for (int nt = 0; nt < 2; ++nt)
          acc[mt][nt] = WMMA_F16(af[mt], bf[nt], acc[mt][nt]);
    }
#if USE_TDM
    buf ^= 1;
#else
    __syncthreads();
#endif
  }

  // C/D fragment: vgpr j -> row (lHi*8 + j), lane lLo -> col.
#pragma unroll
  for (int mt = 0; mt < 4; ++mt)
#pragma unroll
    for (int nt = 0; nt < 2; ++nt)
#pragma unroll
      for (int j = 0; j < 8; ++j) {
        const long r = bM + wm * 64 + mt * 16 + lHi * 8 + j;
        const long c = bN + wn * 32 + nt * 16 + lLo;
        C[r * (long)N + c] = acc[mt][nt][j];
      }
}

// ---------------------------------------------------------------------------
// RoPE for q and k heads; scatters into head-major f16 layouts.
//   qo: [B, H, T, HD]   ko: [B, KV, T, HD]
// ---------------------------------------------------------------------------
__global__ __launch_bounds__(256) void rope_scatter_kernel(
    const float* __restrict__ qkv, const float* __restrict__ fc,
    const float* __restrict__ fs, _Float16* __restrict__ qo,
    _Float16* __restrict__ ko) {
  long i = (long)blockIdx.x * 256 + threadIdx.x;
  const long total = (long)Bq * Tq * (Hq + KVq) * (HDq / 2);
  if (i >= total) return;
  const int d = (int)(i & 63);
  long r = i >> 6;
  const int head = (int)(r % (Hq + KVq));
  r /= (Hq + KVq);
  const int t = (int)(r % Tq);
  const int b = (int)(r / Tq);

  const float* src = qkv + ((long)(b * Tq + t)) * NQKV + head * HDq + 2 * d;
  const float x0 = src[0], x1 = src[1];
  const float c = fc[t * 64 + d], s = fs[t * 64 + d];
  const float o0 = x0 * c - x1 * s;
  const float o1 = x0 * s + x1 * c;
  if (head < Hq) {
    _Float16* dst = qo + (((long)(b * Hq + head)) * Tq + t) * HDq + 2 * d;
    dst[0] = (_Float16)o0; dst[1] = (_Float16)o1;
  } else {
    _Float16* dst = ko + (((long)(b * KVq + (head - Hq))) * Tq + t) * HDq + 2 * d;
    dst[0] = (_Float16)o0; dst[1] = (_Float16)o1;
  }
}

// ---------------------------------------------------------------------------
// V scatter + transpose: qkv fp32 -> vt f16 [B, KV, HD, T]
// ---------------------------------------------------------------------------
__global__ __launch_bounds__(256) void v_scatter_kernel(
    const float* __restrict__ qkv, _Float16* __restrict__ vt) {
  long i = (long)blockIdx.x * 256 + threadIdx.x;
  const long total = (long)Bq * KVq * Tq * HDq;
  if (i >= total) return;
  const int hd = (int)(i & (HDq - 1));
  const int t  = (int)((i >> 7) & (Tq - 1));
  const int kv = (int)((i >> 17) & (KVq - 1));
  const int b  = (int)(i >> 20);
  const float v = qkv[((long)(b * Tq + t)) * NQKV + (Hq + KVq + kv) * HDq + hd];
  vt[(((long)(b * KVq + kv)) * HDq + hd) * Tq + t] = (_Float16)v;
}

// ---------------------------------------------------------------------------
// Flash attention. Grid: B*H*(T/128) blocks of 256 threads (8 waves).
// Wave owns 16 query rows; streams 32-key blocks with online softmax.
// P repack goes through a wave-private LDS slab (no workgroup barriers), so
// each wave clips its key loop to its own causal horizon.
// Y: [B, T, C] f16 (head-major within C) feeds the proj GEMM directly.
// ---------------------------------------------------------------------------
__global__ __launch_bounds__(256) void flash_attn_kernel(
    const _Float16* __restrict__ Q, const _Float16* __restrict__ Kc,
    const _Float16* __restrict__ Vt, _Float16* __restrict__ Y) {
  __shared__ __align__(16) _Float16 pbuf[8][16 * 40];  // per-wave P slab

  const int tid  = threadIdx.x;
  const int wave = tid >> 5;
  const int lane = tid & 31;
  const int lLo  = lane & 15;
  const int lHi  = lane >> 4;

  const int bx   = blockIdx.x;
  const int qblk = bx & 7;
  const int h    = (bx >> 3) & (Hq - 1);
  const int b    = bx >> 8;
  const int g    = h / REPq;
  const int qbase = qblk * 128 + wave * 16;

  const _Float16* qp = Q  + ((long)(b * Hq  + h) * Tq) * HDq;
  const _Float16* kp = Kc + ((long)(b * KVq + g) * Tq) * HDq;
  const _Float16* vp = Vt + ((long)(b * KVq + g) * HDq) * Tq;

  // Q A-fragments (16 x 128 split into 4 K-chunks of 32), resident for the loop.
  v16h qf[4];
#pragma unroll
  for (int kf = 0; kf < 4; ++kf) {
    const _Float16* base = qp + (long)(qbase + lLo) * HDq + kf * 32 + lHi * 8;
    qf[kf] = cat8(*(const v8h*)base, *(const v8h*)(base + 16));
  }

  const v8f zero = {0.f, 0.f, 0.f, 0.f, 0.f, 0.f, 0.f, 0.f};
  v8f acc[8];
#pragma unroll
  for (int nt = 0; nt < 8; ++nt) acc[nt] = zero;
  float mrun[8], lrun[8];
#pragma unroll
  for (int j = 0; j < 8; ++j) { mrun[j] = -3.0e38f; lrun[j] = 0.f; }

  const float scale = 0.088388347648318447f;  // 1/sqrt(128)
  const int kbEnd = (qbase + 15) / 32 + 1;    // per-wave causal horizon

  for (int kb = 0; kb < kbEnd; ++kb) {
    // S = Q * K^T for 32 keys (two 16-col fragments), K-dim = HD = 128.
    v8f s0 = zero, s1 = zero;
#pragma unroll
    for (int kf = 0; kf < 4; ++kf) {
      const _Float16* k0 = kp + (long)(kb * 32 + lLo) * HDq + kf * 32 + lHi * 8;
      s0 = WMMA_F16(qf[kf], cat8(*(const v8h*)k0, *(const v8h*)(k0 + 16)), s0);
      const _Float16* k1 = k0 + 16 * HDq;
      s1 = WMMA_F16(qf[kf], cat8(*(const v8h*)k1, *(const v8h*)(k1 + 16)), s1);
    }

    // Online softmax. Row (lHi*8 + j) lives in one lane-half: xor-reduce over 16.
    float alpha[8];
#pragma unroll
    for (int j = 0; j < 8; ++j) {
      const int row = qbase + lHi * 8 + j;
      const int c0  = kb * 32 + lLo;
      const bool v0 = (c0 <= row);
      const bool v1 = (c0 + 16 <= row);
      const float sa = s0[j] * scale;
      const float sb = s1[j] * scale;
      float a  = v0 ? sa : -3.0e38f;
      float bb = v1 ? sb : -3.0e38f;
      float t = fmaxf(a, bb);
#pragma unroll
      for (int m = 1; m < 16; m <<= 1) t = fmaxf(t, __shfl_xor(t, m, 32));
      const float mn = fmaxf(mrun[j], t);
      const float al = __expf(mrun[j] - mn);
      const float p0 = v0 ? __expf(sa - mn) : 0.f;
      const float p1 = v1 ? __expf(sb - mn) : 0.f;
      float rs = p0 + p1;
#pragma unroll
      for (int m = 1; m < 16; m <<= 1) rs += __shfl_xor(rs, m, 32);
      lrun[j] = lrun[j] * al + rs;
      mrun[j] = mn;
      alpha[j] = al;
      pbuf[wave][(lHi * 8 + j) * 40 + lLo]      = (_Float16)p0;
      pbuf[wave][(lHi * 8 + j) * 40 + 16 + lLo] = (_Float16)p1;
    }
#pragma unroll
    for (int nt = 0; nt < 8; ++nt)
#pragma unroll
      for (int j = 0; j < 8; ++j) acc[nt][j] *= alpha[j];

    wave_lds_fence();  // same-wave DS ops are in-order; just stop reordering

    // P as A-fragment (16x32), V^T rows as B-fragments; 8 WMMAs cover HD=128.
    const v16h pf = cat8(*(const v8h*)&pbuf[wave][lLo * 40 + lHi * 8],
                         *(const v8h*)&pbuf[wave][lLo * 40 + 16 + lHi * 8]);
#pragma unroll
    for (int nt = 0; nt < 8; ++nt) {
      const _Float16* vb = vp + (long)(nt * 16 + lLo) * Tq + kb * 32 + lHi * 8;
      acc[nt] = WMMA_F16(pf, cat8(*(const v8h*)vb, *(const v8h*)(vb + 16)), acc[nt]);
    }
    wave_lds_fence();
  }

  float inv[8];
#pragma unroll
  for (int j = 0; j < 8; ++j) inv[j] = 1.0f / lrun[j];
#pragma unroll
  for (int nt = 0; nt < 8; ++nt)
#pragma unroll
    for (int j = 0; j < 8; ++j) {
      const long row = (long)b * Tq + qbase + lHi * 8 + j;
      Y[row * Cq + h * HDq + nt * 16 + lLo] = (_Float16)(acc[nt][j] * inv[j]);
    }
}

// ---------------------------------------------------------------------------
// Launch sequence.
// ---------------------------------------------------------------------------
extern "C" void kernel_launch(void* const* d_in, const int* in_sizes, int n_in,
                              void* d_out, int out_size, void* d_ws, size_t ws_size,
                              hipStream_t stream) {
  const float* x     = (const float*)d_in[0];
  const float* Wqkv  = (const float*)d_in[1];
  const float* Wproj = (const float*)d_in[2];
  const float* fcos  = (const float*)d_in[3];
  const float* fsin  = (const float*)d_in[4];
  // d_in[5] = mask: unused, causality computed analytically.
  float* out = (float*)d_out;

  // Workspace carve-out (~288 MB total).
  char* wsp = (char*)d_ws;
  auto alloc = [&](size_t bytes) -> char* {
    char* p = wsp;
    wsp += (bytes + 255) & ~(size_t)255;
    return p;
  };
  _Float16* xh     = (_Float16*)alloc((size_t)Bq * Tq * Cq * 2);
  _Float16* wqkvT  = (_Float16*)alloc((size_t)NQKV * Cq * 2);
  _Float16* wprojT = (_Float16*)alloc((size_t)Cq * Cq * 2);
  float*    qkvf   = (float*)   alloc((size_t)Bq * Tq * NQKV * 4);
  _Float16* qh     = (_Float16*)alloc((size_t)Bq * Hq * Tq * HDq * 2);
  _Float16* kh     = (_Float16*)alloc((size_t)Bq * KVq * Tq * HDq * 2);
  _Float16* vth    = (_Float16*)alloc((size_t)Bq * KVq * HDq * Tq * 2);
  _Float16* yh     = (_Float16*)alloc((size_t)Bq * Tq * Cq * 2);

  // 1) x -> f16
  {
    const long n = (long)Bq * Tq * Cq;
    f32_to_f16_kernel<<<dim3((unsigned)(n / 4 / 256)), 256, 0, stream>>>(x, xh, n);
  }
  // 2) W^T -> f16 (B operand is N-major for the WMMA fragment load pattern)
  transpose_f32_f16_kernel<<<dim3(NQKV / 32, Cq / 32), dim3(32, 8), 0, stream>>>(Wqkv, wqkvT, Cq, NQKV);
  transpose_f32_f16_kernel<<<dim3(Cq / 32, Cq / 32), dim3(32, 8), 0, stream>>>(Wproj, wprojT, Cq, Cq);
  // 3) qkv = x @ Wqkv
  gemm_f16_wmma<<<dim3(NQKV / 128, (Bq * Tq) / 128), 256, 0, stream>>>(
      xh, wqkvT, qkvf, Bq * Tq, NQKV, Cq);
  // 4) RoPE(q,k) + v transpose, scatter to head-major f16
  {
    const long n = (long)Bq * Tq * (Hq + KVq) * (HDq / 2);
    rope_scatter_kernel<<<dim3((unsigned)((n + 255) / 256)), 256, 0, stream>>>(
        qkvf, fcos, fsin, qh, kh);
  }
  {
    const long n = (long)Bq * KVq * Tq * HDq;
    v_scatter_kernel<<<dim3((unsigned)((n + 255) / 256)), 256, 0, stream>>>(qkvf, vth);
  }
  // 5) flash attention -> yh [B,T,C] f16
  flash_attn_kernel<<<dim3(Bq * Hq * (Tq / 128)), 256, 0, stream>>>(qh, kh, vth, yh);
  // 6) out = y @ Wproj (fp32 output)
  gemm_f16_wmma<<<dim3(Cq / 128, (Bq * Tq) / 128), 256, 0, stream>>>(
      yh, wprojT, out, Bq * Tq, Cq, Cq);
}
